// Decode_78572131713670
// MI455X (gfx1250) — matrix operand order repaired and checked by reading, not compiled, not run on hardware
//
#include <hip/hip_runtime.h>
#include <math.h>

// ---------------- problem constants ----------------
#define B_   128
#define T_   512
#define H_   400
#define M_   (T_ * B_)      // 65536 rows (time-major: m = t*B + b)
#define KZ   928            // fc input K: 920 padded to 29*32
#define NZ   416            // z width: 400 padded to 13*32 (also K of next GEMM)
#define NG   1600           // gates width (4*H), = 50 tiles of 32
#define NC   64             // combine width: 60 padded to 64
#define NEGV (-1e30f)

// ---------------- CDNA5 WMMA types ----------------
typedef __attribute__((ext_vector_type(16))) __bf16    v16bf;
typedef __attribute__((ext_vector_type(8)))  float     v8f;
typedef __attribute__((ext_vector_type(4)))  unsigned  u32x4;

union FragU { v16bf v; u32x4 q[2]; };

__device__ __forceinline__ unsigned short f2bf(float x) {
    unsigned u = __float_as_uint(x);
    return (unsigned short)((u + 0x7FFFu + ((u >> 16) & 1u)) >> 16);  // RNE
}
__device__ __forceinline__ float bf2f(unsigned short h) {
    return __uint_as_float(((unsigned)h) << 16);
}
__device__ __forceinline__ float sigm(float x) {
    return 1.f / (1.f + __expf(-x));
}

// A fragment: 16x32 bf16 tile of A[m, k] (row-major, ld in elements).
// lanes 0-15: m = base+lane, K chunks [0..7],[16..23]; lanes 16-31: K [8..15],[24..31]
__device__ __forceinline__ v16bf load_fragA(const unsigned short* __restrict__ A,
                                            int ld, int mBase, int kBase) {
    int lane = threadIdx.x & 31;
    int m    = mBase + (lane & 15);
    int koff = kBase + ((lane < 16) ? 0 : 8);
    const u32x4* p = reinterpret_cast<const u32x4*>(A + (long)m * ld + koff);
    FragU f; f.q[0] = p[0]; f.q[1] = p[2];   // +16 bf16 for second K block
    return f.v;
}
// B fragment: 32x16 tile where B[k,n] = W[n,k] (weights row-major [N,K]).
// lanes 0-15: n = base+lane, K chunks [0..7],[16..23]; lanes 16-31: K [8..15],[24..31]
__device__ __forceinline__ v16bf load_fragB(const unsigned short* __restrict__ W,
                                            int ld, int nBase, int kBase) {
    int lane = threadIdx.x & 31;
    int n    = nBase + (lane & 15);
    int koff = kBase + ((lane < 16) ? 0 : 8);
    const u32x4* p = reinterpret_cast<const u32x4*>(W + (long)n * ld + koff);
    FragU f; f.q[0] = p[0]; f.q[1] = p[2];
    return f.v;
}
__device__ __forceinline__ v8f wmma_bf16(v16bf a, v16bf b, v8f c) {
    return __builtin_amdgcn_wmma_f32_16x16x32_bf16(false, a, false, b,
                                                   (short)0, c, false, false);
}

// ---------------- pack fp32 -> zero-padded bf16 [rowsP, colsP] ----------------
__global__ void pack_bf16_kernel(const float* __restrict__ src, unsigned short* __restrict__ dst,
                                 int rows, int cols, int rowsP, int colsP,
                                 int srcLd, int srcOff) {
    long total = (long)rowsP * colsP;
    for (long i = (long)blockIdx.x * blockDim.x + threadIdx.x; i < total;
         i += (long)gridDim.x * blockDim.x) {
        int r = (int)(i / colsP), c = (int)(i % colsP);
        float v = (r < rows && c < cols) ? src[(long)r * srcLd + srcOff + c] : 0.f;
        dst[i] = f2bf(v);
    }
}

// ---------------- build Zin rows: [mean(800) | wl_emb(20) | pos_emb(100) | pad(8)] ----------------
__global__ void build_zin_kernel(const float* __restrict__ enc,        // [B,T,2H]
                                 const int* __restrict__ posv,         // [B,T]
                                 const int* __restrict__ wlv,          // [B,T]
                                 const float* __restrict__ posE,       // [32,100]
                                 const float* __restrict__ wlE,        // [8,20]
                                 unsigned short* __restrict__ Zin) {   // [M_, KZ]
    int m = blockIdx.x;                 // m = t*B + b
    int t = m / B_, b = m % B_;
    int wl  = wlv[(long)b * T_ + t];
    int len = wl < 1 ? 1 : (wl > 6 ? 6 : wl);
    int tmax = t > 1 ? t : 1;
    if (len > tmax) len = tmax;
    float inv = 1.f / (float)len;
    int pos = posv[(long)b * T_ + t];
    unsigned short* dst = Zin + (long)m * KZ;
    for (int k = threadIdx.x; k < KZ; k += blockDim.x) {
        float v;
        if (k < 800) {                      // window mean over <=6 previous rows
            float s = 0.f;
            for (int j = t - len; j < t; ++j)
                if (j >= 0) s += enc[((long)b * T_ + j) * 800 + k];
            v = s * inv;                    // t==0 row is zeroed later anyway
        } else if (k < 820) v = wlE[len * 20 + (k - 800)];
        else if (k < 920)   v = posE[pos * 100 + (k - 820)];
        else                v = 0.f;
        dst[k] = f2bf(v);
    }
}

// ---------------- generic 32x32-per-wave bf16 WMMA GEMM -------------------------
// C[m,n] = act(sum_k A[m,k]*W[n,k] + bias)
// MODE 0: out = bf16(tanh(acc + bias0[n<nBiasValid]))   (Z)
// MODE 1: out = bf16(acc + bias0[n] + bias1[n])         (Gpre)
template <int MODE>
__global__ void gemm_wmma_kernel(const unsigned short* __restrict__ A,
                                 const unsigned short* __restrict__ W,
                                 unsigned short* __restrict__ outB,
                                 const float* __restrict__ bias0,
                                 const float* __restrict__ bias1,
                                 int M, int Kpad, int Npad, int nBiasValid) {
    int wid  = threadIdx.x >> 5;
    int lane = threadIdx.x & 31;
    int mT2  = M >> 5;                       // 32-row blocks
    int nT2  = Npad >> 5;                    // 32-col blocks
    long tile = (long)blockIdx.x * (blockDim.x >> 5) + wid;
    if (tile >= (long)mT2 * nT2) return;
    int tM16 = (int)(tile % mT2) * 2;        // in 16-units
    int tN16 = (int)(tile / mT2) * 2;

    v8f acc[2][2] = {};
    for (int kk = 0; kk < Kpad; kk += 32) {
        v16bf a0 = load_fragA(A, Kpad, tM16 * 16, kk);
        v16bf a1 = load_fragA(A, Kpad, (tM16 + 1) * 16, kk);
        v16bf b0 = load_fragB(W, Kpad, tN16 * 16, kk);
        v16bf b1 = load_fragB(W, Kpad, (tN16 + 1) * 16, kk);
        acc[0][0] = wmma_bf16(a0, b0, acc[0][0]);
        acc[0][1] = wmma_bf16(a0, b1, acc[0][1]);
        acc[1][0] = wmma_bf16(a1, b0, acc[1][0]);
        acc[1][1] = wmma_bf16(a1, b1, acc[1][1]);
    }
    int lane15 = lane & 15;
    int rowSel = (lane < 16) ? 0 : 8;
    for (int i = 0; i < 2; ++i) {
        int mBase = (tM16 + i) * 16 + rowSel;
        for (int j = 0; j < 2; ++j) {
            int n = (tN16 + j) * 16 + lane15;
            float bias;
            if constexpr (MODE == 0) bias = (n < nBiasValid) ? bias0[n] : 0.f;
            else                     bias = bias0[n] + bias1[n];
            for (int r = 0; r < 8; ++r) {
                float v = acc[i][j][r] + bias;
                if constexpr (MODE == 0) v = tanhf(v);
                outB[(long)(mBase + r) * Npad + n] = f2bf(v);
            }
        }
    }
}

// ---------------- Oenc[m,n] = sum_{k<800} enc[b,t,k] * combine_w[n, 400+k] (fp32) ----------------
__global__ void oenc_kernel(const float* __restrict__ enc, const float* __restrict__ cw,
                            float* __restrict__ Oenc) {
    __shared__ float srow[4][800];
    int mBase = blockIdx.x * 4;
    for (int i = threadIdx.x; i < 4 * 800; i += blockDim.x) {
        int mi = i / 800, k = i % 800;
        int m = mBase + mi; int t = m / B_, b = m % B_;
        srow[mi][k] = enc[((long)b * T_ + t) * 800 + k];
    }
    __syncthreads();
    int mi = threadIdx.x >> 6;   // 0..3
    int n  = threadIdx.x & 63;   // 0..63
    float acc = 0.f;
    if (n < 60) {
        const float* w = cw + (long)n * 1200 + 400;
        for (int k = 0; k < 800; ++k) acc = fmaf(srow[mi][k], w[k], acc);
    }
    Oenc[(long)(mBase + mi) * NC + n] = acc;
}

// ---------------- small init / zero kernels ----------------
__global__ void zero_u16_kernel(unsigned short* p, int n) {
    int i = blockIdx.x * blockDim.x + threadIdx.x;
    if (i < n) p[i] = 0;
}
__global__ void seq_init_kernel(unsigned short* hB, float* cSt, unsigned* bar) {
    int i = blockIdx.x * blockDim.x + threadIdx.x;
    if (i < B_ * NZ)  hB[i] = 0;
    if (i < B_ * H_)  cSt[i] = 0.f;
    if (i < 8)        bar[i] = 0u;
}

// ---------------- grid-wide barrier (persistent grid, all blocks resident) ----------------
__device__ __forceinline__ void gsync(volatile unsigned* cnt, volatile unsigned* gen,
                                      unsigned nb) {
    __syncthreads();
    if (threadIdx.x == 0) {
        __threadfence();
        unsigned g = *gen;
        unsigned a = atomicAdd((unsigned*)cnt, 1u) + 1u;
        if (a == nb) {
            *cnt = 0u;
            __threadfence();
            atomicAdd((unsigned*)gen, 1u);
        } else {
            while (*gen == g) { __builtin_amdgcn_s_sleep(2); }
        }
        __threadfence();
    }
    __syncthreads();
}

// ---------------- sequential recurrence: 512 steps, persistent grid ----------------
#define SEQ_BLOCKS 128
__global__ void __launch_bounds__(256, 1)
decode_seq_kernel(const unsigned short* __restrict__ Gpre,    // [M_, NG] bf16
                  const unsigned short* __restrict__ whhB,    // [NG, NZ] bf16
                  const unsigned short* __restrict__ combHB,  // [NC, NZ] bf16
                  const float* __restrict__ Oenc,             // [M_, NC]
                  const unsigned char* __restrict__ mask,     // [B_, T_]
                  unsigned short* __restrict__ hB,            // [B_, NZ] bf16 (pads zero)
                  float* __restrict__ cSt,                    // [B_, H_]
                  float* __restrict__ gates,                  // [B_, NG]
                  unsigned* __restrict__ bar,
                  float* __restrict__ out) {                  // [T_, B_, 60]
    const unsigned nb = gridDim.x;
    int wid  = threadIdx.x >> 5;
    int lane = threadIdx.x & 31;
    int gw   = blockIdx.x * (blockDim.x >> 5) + wid;   // global wave id
    int lane15 = lane & 15;
    int rowSel = (lane < 16) ? 0 : 8;

    for (int t = 0; t < T_; ++t) {
        // ---- Phase A: gates = Gpre[t] + h @ w_hh^T   (4 x 50 blocks of 32x32, waves 0..199)
        if (gw < (B_ / 32) * (NG / 32)) {
            int tM16 = (gw & 3) * 2;         // 0,2,4,6
            int tN16 = (gw >> 2) * 2;        // 0..98 even
            v8f acc[2][2];
            for (int i = 0; i < 2; ++i) {
                int mBase = (tM16 + i) * 16 + rowSel;
                for (int j = 0; j < 2; ++j) {
                    int n = (tN16 + j) * 16 + lane15;
                    for (int r = 0; r < 8; ++r)
                        acc[i][j][r] = bf2f(Gpre[((long)t * B_ + mBase + r) * NG + n]);
                }
            }
            for (int kk = 0; kk < NZ; kk += 32) {
                v16bf a0 = load_fragA(hB, NZ, tM16 * 16, kk);
                v16bf a1 = load_fragA(hB, NZ, (tM16 + 1) * 16, kk);
                v16bf b0 = load_fragB(whhB, NZ, tN16 * 16, kk);
                v16bf b1 = load_fragB(whhB, NZ, (tN16 + 1) * 16, kk);
                acc[0][0] = wmma_bf16(a0, b0, acc[0][0]);
                acc[0][1] = wmma_bf16(a0, b1, acc[0][1]);
                acc[1][0] = wmma_bf16(a1, b0, acc[1][0]);
                acc[1][1] = wmma_bf16(a1, b1, acc[1][1]);
            }
            for (int i = 0; i < 2; ++i) {
                int mBase = (tM16 + i) * 16 + rowSel;
                for (int j = 0; j < 2; ++j) {
                    int n = (tN16 + j) * 16 + lane15;
                    for (int r = 0; r < 8; ++r)
                        gates[(long)(mBase + r) * NG + n] = acc[i][j][r];
                }
            }
            // prefetch next timestep's Gpre tile (hides HBM latency on serial path)
            if (t + 1 < T_) {
                for (int i = 0; i < 2; ++i)
                    for (int j = 0; j < 2; ++j)
                        __builtin_prefetch(
                            &Gpre[((long)(t + 1) * B_ + (tM16 + i) * 16 + rowSel) * NG +
                                  (tN16 + j) * 16 + lane15], 0, 1);
            }
        }
        gsync(&bar[0], &bar[1], nb);

        // ---- Phase B: LSTM pointwise, update c / h (gate order i,f,g,o)
        for (int i = blockIdx.x * blockDim.x + threadIdx.x; i < B_ * H_;
             i += (int)nb * blockDim.x) {
            int b = i / H_, j = i % H_;
            const float* g = gates + (long)b * NG;
            float gi = g[j], gf = g[H_ + j], gg = g[2 * H_ + j], go = g[3 * H_ + j];
            float cn = sigm(gf) * cSt[i] + sigm(gi) * tanhf(gg);
            float hn = sigm(go) * tanhf(cn);
            cSt[i] = cn;
            hB[(long)b * NZ + j] = f2bf(hn);     // pad cols [400,416) stay zero
        }
        gsync(&bar[0], &bar[1], nb);

        // ---- Phase C: out = h @ combine_h^T + Oenc[t], masking (8 x 4 tiles, waves 0..31)
        if (gw < (B_ / 16) * (NC / 16)) {
            int tM = gw & 7;
            int tN = gw >> 3;        // 0..3
            v8f acc = {0.f, 0.f, 0.f, 0.f, 0.f, 0.f, 0.f, 0.f};
            for (int kk = 0; kk < NZ; kk += 32) {
                v16bf a = load_fragA(hB, NZ, tM * 16, kk);
                v16bf b = load_fragB(combHB, NZ, tN * 16, kk);
                acc = wmma_bf16(a, b, acc);
            }
            int n     = tN * 16 + lane15;
            int mBase = tM * 16 + rowSel;
            for (int r = 0; r < 8; ++r) {
                int b = mBase + r;
                float v = acc[r] + Oenc[((long)t * B_ + b) * NC + n];
                if (n == 0) v = NEGV;                      // PAD_ID always masked
                if (t == 0 && n == 1) v = NEGV;            // APP_ID masked at t==0
                if (!mask[(long)b * T_ + t]) v = NEGV;     // sequence mask
                if (n < 60) out[((long)t * B_ + b) * 60 + n] = v;
            }
            if (t + 1 < T_)                                // prefetch next Oenc slab
                __builtin_prefetch(&Oenc[((long)(t + 1) * B_ + mBase) * NC + n], 0, 1);
        }
        gsync(&bar[0], &bar[1], nb);
    }
}

// ---------------- host launcher ----------------
extern "C" void kernel_launch(void* const* d_in, const int* in_sizes, int n_in,
                              void* d_out, int out_size, void* d_ws, size_t ws_size,
                              hipStream_t stream) {
    (void)in_sizes; (void)n_in; (void)out_size; (void)ws_size;
    const float*         enc   = (const float*)d_in[0];
    const unsigned char* mask  = (const unsigned char*)d_in[1];
    const int*           posv  = (const int*)d_in[2];
    const int*           wlv   = (const int*)d_in[3];
    const float*         posE  = (const float*)d_in[4];
    const float*         wlE   = (const float*)d_in[5];
    const float*         fc_w  = (const float*)d_in[6];
    const float*         fc_b  = (const float*)d_in[7];
    const float*         w_ih  = (const float*)d_in[8];
    const float*         w_hh  = (const float*)d_in[9];
    const float*         b_ih  = (const float*)d_in[10];
    const float*         b_hh  = (const float*)d_in[11];
    const float*         comb  = (const float*)d_in[12];
    float*               out   = (float*)d_out;

    // workspace carve (256B aligned); Zin aliased under Gpre (Zin dead before Gpre written)
    char* ws = (char*)d_ws;
    size_t off = 0;
    auto carve = [&](size_t bytes) -> void* {
        void* p = ws + off; off += (bytes + 255) & ~(size_t)255; return p;
    };
    unsigned short* Gpre   = (unsigned short*)carve((size_t)M_ * NG * 2);  // 210 MB
    unsigned short* Zin    = Gpre;                                          // 121.6 MB alias
    unsigned short* Z      = (unsigned short*)carve((size_t)M_ * NZ * 2);  // 54.5 MB
    float*          Oenc   = (float*)carve((size_t)M_ * NC * 4);           // 16.8 MB
    unsigned short* fcB    = (unsigned short*)carve((size_t)NZ * KZ * 2);  // [416,928]
    unsigned short* wihB   = (unsigned short*)carve((size_t)NG * NZ * 2);  // [1600,416]
    unsigned short* whhB   = (unsigned short*)carve((size_t)NG * NZ * 2);  // [1600,416]
    unsigned short* combHB = (unsigned short*)carve((size_t)NC * NZ * 2);  // [64,416]
    unsigned short* hB     = (unsigned short*)carve((size_t)B_ * NZ * 2);
    float*          cSt    = (float*)carve((size_t)B_ * H_ * 4);
    float*          gatesF = (float*)carve((size_t)B_ * NG * 4);
    unsigned*       bar    = (unsigned*)carve(256);

    auto gsz = [](long total) { long b = (total + 255) / 256; return (int)(b > 8192 ? 8192 : b); };

    // 1) pack weights to padded bf16 [N,K]
    pack_bf16_kernel<<<gsz((long)NZ * KZ), 256, 0, stream>>>(fc_w, fcB, 400, 920, NZ, KZ, 920, 0);
    pack_bf16_kernel<<<gsz((long)NG * NZ), 256, 0, stream>>>(w_ih, wihB, NG, 400, NG, NZ, 400, 0);
    pack_bf16_kernel<<<gsz((long)NG * NZ), 256, 0, stream>>>(w_hh, whhB, NG, 400, NG, NZ, 400, 0);
    pack_bf16_kernel<<<gsz((long)NC * NZ), 256, 0, stream>>>(comb, combHB, 60, 400, NC, NZ, 1200, 0);

    // 2) build Zin rows (windowed mean + embeddings), bf16
    build_zin_kernel<<<M_, 256, 0, stream>>>(enc, posv, wlv, posE, wlE, Zin);

    // 3) Z = tanh(Zin @ fc_w^T + fc_b)       M=65536, K=928, N=416 (32x32 blocks: 2048x13)
    {
        long tiles = (long)(M_ / 32) * (NZ / 32);
        gemm_wmma_kernel<0><<<(int)((tiles + 7) / 8), 256, 0, stream>>>(
            Zin, fcB, Z, fc_b, nullptr, M_, KZ, NZ, 400);
    }
    // 4) zero Z rows for t==0 (first 128 rows)
    zero_u16_kernel<<<(B_ * NZ + 255) / 256, 256, 0, stream>>>(Z, B_ * NZ);

    // 5) Gpre = Z @ w_ih^T + b_ih + b_hh     M=65536, K=416, N=1600 (32x32 blocks: 2048x50)
    {
        long tiles = (long)(M_ / 32) * (NG / 32);
        gemm_wmma_kernel<1><<<(int)((tiles + 7) / 8), 256, 0, stream>>>(
            Z, wihB, Gpre, b_ih, b_hh, M_, NZ, NG, NG);
    }

    // 6) Oenc = enc @ combine_w[:, H:]^T (fp32)
    oenc_kernel<<<M_ / 4, 256, 0, stream>>>(enc, comb, Oenc);

    // 7) init recurrent state + barrier
    seq_init_kernel<<<(B_ * NZ + 255) / 256, 256, 0, stream>>>(hB, cSt, bar);

    // 8) sequential LSTM recurrence, persistent grid with global barriers
    decode_seq_kernel<<<SEQ_BLOCKS, 256, 0, stream>>>(
        Gpre, whhB, combHB, Oenc, mask, hB, cSt, gatesF, bar, out);
}
